// MultiHeadAttention_1486058684576
// MI455X (gfx1250) — matrix operand order repaired and checked by reading, not compiled, not run on hardware
//
#include <hip/hip_runtime.h>
#include <hip/hip_bf16.h>

typedef __attribute__((ext_vector_type(16))) _Float16 v16h;
typedef __attribute__((ext_vector_type(8)))  float    v8f;
typedef __attribute__((ext_vector_type(4)))  int      v4i;

#define B_  2
#define S_  2048
#define D_  1024
#define H_  16
#define DP_ 64
#define M_  (B_*S_)   // 4096 rows

// dynamic LDS for fused attention:
// sS  : 16*2048 f32 = 131072 B (score/prob strip, reused for partial-ctx reduce)
// sRed: 16*16   f32 =   1024 B
// sQ  : 16*64   f16 =   2048 B
// sV  : 2*64*64 f16 =  16384 B (double-buffered V tile, un-transposed)
#define ATTN_SMEM_BYTES (131072 + 1024 + 2048 + 16384)   // 150528

// CDNA5 async global->LDS path (guarded; falls back to reg-staged copies)
#if defined(__gfx1250__) && __has_builtin(__builtin_amdgcn_global_load_async_to_lds_b128)
#define HAS_ASYNC_LDS 1
#else
#define HAS_ASYNC_LDS 0
#endif

union U16x8 { uint4 u; _Float16 h[8]; };

__device__ __forceinline__ void wait_async0() {
#if defined(__gfx1250__) && __has_builtin(__builtin_amdgcn_s_wait_asynccnt)
  __builtin_amdgcn_s_wait_asynccnt(0);
#else
  asm volatile("s_wait_asynccnt 0" ::: "memory");
#endif
}

#if HAS_ASYNC_LDS
__device__ __forceinline__ void async_copy16(const _Float16* g, _Float16* l) {
  // prototype (from hipcc diagnostic): param0 = v4i __device__* (AS1, non-const)
  __builtin_amdgcn_global_load_async_to_lds_b128(
      (__attribute__((address_space(1))) v4i*)g,
      (__attribute__((address_space(3))) v4i*)l, 0, 0);
}
#endif

__device__ __forceinline__ void load_pair_h(const _Float16* p, v16h& f, int j) {
  union { unsigned u; _Float16 h[2]; } t;
  t.u = *(const unsigned*)p;          // one b32 load of two contiguous f16
  f[2*j]   = t.h[0];
  f[2*j+1] = t.h[1];
}

// ---------------------------------------------------------------- convert
__global__ void cvt_f32_f16_kernel(const float* __restrict__ src,
                                   _Float16* __restrict__ dst, int n) {
  int i = blockIdx.x * blockDim.x + threadIdx.x;
  if (i < n) dst[i] = (_Float16)src[i];
}

// ---------------------------------------------------------------- GEMM
// C[M,N] = A[M,K] @ W[K,N] + bias.  Block tile 128x64, 8 waves of 32x32.
// Double-buffered LDS staging pipelined against WMMA.
// MODE 0: store f16 split-head [B,H,S,DP];  MODE 1: store f32 row-major.
template<int MODE>
__global__ void __launch_bounds__(256)
gemm_f16_wmma_kernel(const _Float16* __restrict__ A,
                     const _Float16* __restrict__ W,
                     const float* __restrict__ bias,
                     _Float16* __restrict__ outH,
                     float* __restrict__ outF,
                     int M, int N, int K)
{
  __shared__ __attribute__((aligned(16))) _Float16 sA[2*128*32];  // 16 KB
  __shared__ __attribute__((aligned(16))) _Float16 sB[2*32*64];   //  8 KB

  const int tid  = threadIdx.x;
  const int wave = tid >> 5;
  const int lane = tid & 31;
  const int lh   = lane >> 4;
  const int l16  = lane & 15;
  const int m0 = blockIdx.y * 128;
  const int n0 = blockIdx.x * 64;
  const int mw = (wave >> 1) * 32;
  const int nw = (wave & 1) * 32;

  // staging geometry (per buffer): A = 512 uint4 (2/thread), B = 256 (1/thread)
  const int ar0 = tid >> 2,         as0 = (tid & 3) * 8;
  const int ar1 = (tid + 256) >> 2, as1 = ((tid + 256) & 3) * 8;
  const int bk  = tid >> 3,         bs  = (tid & 7) * 8;

  v8f acc[2][2];
  const v8f vzero = {0.f,0.f,0.f,0.f,0.f,0.f,0.f,0.f};
  acc[0][0] = vzero; acc[0][1] = vzero; acc[1][0] = vzero; acc[1][1] = vzero;

  const int NITK = K / 32;   // 32

  // prologue: stage tile 0 into buffer 0
  {
#if HAS_ASYNC_LDS
    async_copy16(A + (size_t)(m0 + ar0)*K + as0, sA + ar0*32 + as0);
    async_copy16(A + (size_t)(m0 + ar1)*K + as1, sA + ar1*32 + as1);
    async_copy16(W + (size_t)bk*N + n0 + bs,     sB + bk*64 + bs);
    wait_async0();
#else
    *(uint4*)(sA + ar0*32 + as0) = *(const uint4*)(A + (size_t)(m0 + ar0)*K + as0);
    *(uint4*)(sA + ar1*32 + as1) = *(const uint4*)(A + (size_t)(m0 + ar1)*K + as1);
    *(uint4*)(sB + bk*64 + bs)   = *(const uint4*)(W + (size_t)bk*N + n0 + bs);
#endif
  }
  __syncthreads();

  for (int it = 0; it < NITK; ++it) {
    const int cur = it & 1, nxt = cur ^ 1;
    const bool hs = (it + 1 < NITK);
    const int kk2 = (it + 1) * 32;

#if HAS_ASYNC_LDS
    if (hs) {   // issue async staging for next tile, overlaps with WMMA below
      async_copy16(A + (size_t)(m0 + ar0)*K + kk2 + as0, sA + nxt*4096 + ar0*32 + as0);
      async_copy16(A + (size_t)(m0 + ar1)*K + kk2 + as1, sA + nxt*4096 + ar1*32 + as1);
      async_copy16(W + (size_t)(kk2 + bk)*N + n0 + bs,   sB + nxt*2048 + bk*64 + bs);
    }
#else
    U16x8 ta0, ta1, tb0;
    if (hs) {   // loads issued early; latency overlaps with WMMA below
      ta0.u = *(const uint4*)(A + (size_t)(m0 + ar0)*K + kk2 + as0);
      ta1.u = *(const uint4*)(A + (size_t)(m0 + ar1)*K + kk2 + as1);
      tb0.u = *(const uint4*)(W + (size_t)(kk2 + bk)*N + n0 + bs);
    }
#endif

    const _Float16* sAc = sA + cur*4096;
    const _Float16* sBc = sB + cur*2048;
    v16h aF[2], bF[2];
    #pragma unroll
    for (int t = 0; t < 2; ++t) {
      int m = mw + t*16 + l16;
      #pragma unroll
      for (int j = 0; j < 8; ++j) {
        int k0 = lh*16 + 2*j;                      // A in contiguous k = 16*lh + s order (same as B)
        load_pair_h(&sAc[m*32 + k0], aF[t], j);
      }
    }
    #pragma unroll
    for (int t = 0; t < 2; ++t) {
      int n = nw + t*16 + l16;
      #pragma unroll
      for (int j = 0; j < 8; ++j) {
        int k0 = lh*16 + 2*j;                      // B in contiguous k = 16*lh + s order (same as A)
        bF[t][2*j]   = sBc[k0*64 + n];
        bF[t][2*j+1] = sBc[(k0 + 1)*64 + n];
      }
    }
    #pragma unroll
    for (int i = 0; i < 2; ++i)
      #pragma unroll
      for (int j = 0; j < 2; ++j)
        acc[i][j] = __builtin_amdgcn_wmma_f32_16x16x32_f16(
            false, aF[i], false, bF[j], (short)0, acc[i][j], false, false);

#if HAS_ASYNC_LDS
    if (hs) wait_async0();
#else
    if (hs) {
      *(uint4*)(sA + nxt*4096 + ar0*32 + as0) = ta0.u;
      *(uint4*)(sA + nxt*4096 + ar1*32 + as1) = ta1.u;
      *(uint4*)(sB + nxt*2048 + bk*64 + bs)   = tb0.u;
    }
#endif
    __syncthreads();
  }

  // epilogue
  #pragma unroll
  for (int i = 0; i < 2; ++i) {
    #pragma unroll
    for (int j = 0; j < 2; ++j) {
      const int ng = n0 + nw + j*16 + l16;
      const float bv = bias[ng];
      #pragma unroll
      for (int r = 0; r < 8; ++r) {
        int mg = m0 + mw + i*16 + lh*8 + r;
        float val = acc[i][j][r] + bv;
        if (MODE == 1) {
          outF[(size_t)mg*N + ng] = val;
        } else {
          int bidx = mg >> 11, s = mg & (S_ - 1);
          int h = ng >> 6, d = ng & (DP_ - 1);
          outH[(((size_t)bidx*H_ + h)*S_ + s)*DP_ + d] = (_Float16)val;
        }
      }
    }
  }
}

// ---------------------------------------------------------------- fused attention
// One block per (b,h, 16 query rows). 256 threads = 8 waves.
// Phase 1: scores = QK^T/8 into LDS (8 waves over key blocks, B-frags from
//          row-major K directly: depth pairs are contiguous). Prefetch next tile.
// Phase 2: softmax over 2048 keys in LDS, single-pass write of probs to attn out.
// Phase 3: ctx = P @ V. 8 waves: d-tile = wave&3, key-half = wave>>2.
//          Double-buffered un-transposed V tile, partial-ctx reduce via LDS.
__global__ void __launch_bounds__(256)
attn_fused_kernel(const _Float16* __restrict__ Qp,
                  const _Float16* __restrict__ Kp,
                  const _Float16* __restrict__ Vp,
                  float* __restrict__ attnOut,
                  _Float16* __restrict__ ctxOut)
{
  extern __shared__ __attribute__((aligned(16))) char smem[];
  float*    sS   = (float*)smem;
  float*    sRed = (float*)(smem + 131072);
  _Float16* sQ   = (_Float16*)(smem + 131072 + 1024);
  _Float16* sV   = (_Float16*)(smem + 131072 + 1024 + 2048);  // [2][64][64]

  const int tid  = threadIdx.x;
  const int wave = tid >> 5;
  const int lane = tid & 31;
  const int lh   = lane >> 4;
  const int l16  = lane & 15;

  const int blk = blockIdx.x;         // bh * 128 + qb
  const int qb  = blk & 127;
  const int bh  = blk >> 7;           // b*16 + h

  const _Float16* Qb = Qp + ((size_t)bh*S_ + qb*16)*DP_;
  const _Float16* Kb = Kp + (size_t)bh*S_*DP_;
  const _Float16* Vb = Vp + (size_t)bh*S_*DP_;

  // stage Q tile 16x64 f16 = 128 uint4
  if (tid < 128) {
#if HAS_ASYNC_LDS
    async_copy16(Qb + tid*8, sQ + tid*8);
#else
    ((uint4*)sQ)[tid] = ((const uint4*)Qb)[tid];
#endif
  }
#if HAS_ASYNC_LDS
  wait_async0();
#endif
  __syncthreads();

  const float scale = 0.125f;         // 1/sqrt(64)

  // ---- Phase 1: scores
  for (int kb = wave; kb < 128; kb += 8) {
    const int key = kb*16 + l16;
    if (kb + 8 < 128)   // prefetch this lane's next key row (global_prefetch_b8)
      __builtin_prefetch((const void*)(Kb + (size_t)(key + 128)*DP_), 0, 0);
    v8f c = {0.f,0.f,0.f,0.f,0.f,0.f,0.f,0.f};
    #pragma unroll
    for (int kc = 0; kc < 2; ++kc) {
      v16h a, b;
      #pragma unroll
      for (int j = 0; j < 8; ++j) {   // A = Q rows, depth chunk
        int k0 = kc*32 + lh*16 + 2*j;
        load_pair_h(&sQ[l16*DP_ + k0], a, j);
      }
      #pragma unroll
      for (int j = 0; j < 8; ++j) {   // B = K^T: pairs contiguous in K row
        int k0 = kc*32 + lh*16 + 2*j;
        load_pair_h(Kb + (size_t)key*DP_ + k0, b, j);
      }
      c = __builtin_amdgcn_wmma_f32_16x16x32_f16(
          false, a, false, b, (short)0, c, false, false);
    }
    #pragma unroll
    for (int r = 0; r < 8; ++r) {
      int m = lh*8 + r;
      sS[m*S_ + kb*16 + l16] = c[r] * scale;
    }
  }
  __syncthreads();

  // ---- Phase 2: softmax (16 rows x 16 threads)
  const int row  = tid >> 4;
  const int col0 = tid & 15;
  float mx = -3.0e38f;
  for (int c = col0; c < S_; c += 16) mx = fmaxf(mx, sS[row*S_ + c]);
  sRed[row*16 + col0] = mx;
  __syncthreads();
  if (col0 == 0) {
    float m2 = sRed[row*16];
    #pragma unroll
    for (int i = 1; i < 16; ++i) m2 = fmaxf(m2, sRed[row*16 + i]);
    sRed[row*16] = m2;
  }
  __syncthreads();
  const float rmax = sRed[row*16];
  float sum = 0.f;
  for (int c = col0; c < S_; c += 16) {
    float e = __expf(sS[row*S_ + c] - rmax);
    sS[row*S_ + c] = e;
    sum += e;
  }
  __syncthreads();
  sRed[row*16 + col0] = sum;
  __syncthreads();
  if (col0 == 0) {
    float t = 0.f;
    #pragma unroll
    for (int i = 0; i < 16; ++i) t += sRed[row*16 + i];
    sRed[row*16] = t;
  }
  __syncthreads();
  const float inv = 1.0f / sRed[row*16];
  float* attnRow = attnOut + ((size_t)bh*S_ + qb*16 + row)*S_;
  for (int c = col0; c < S_; c += 16) {
    float p = sS[row*S_ + c] * inv;
    sS[row*S_ + c] = p;                // keep probs for PV
    attnRow[c] = p;                    // single pass over the 537MB output
  }
  __syncthreads();

  // ---- Phase 3: ctx = P @ V, 8 active waves, double-buffered V staging
  const int dt  = wave & 3;           // d-tile (16 cols)
  const int kh2 = wave >> 2;          // key-half within the 64-key step
  // staging geometry per 64-key tile (4096 f16 = 512 uint4, 2 per thread)
  const int vk0 = tid >> 2;
  const int c1  = tid + 256;
  const int vk1 = c1 >> 2;
  const int vsA = (tid & 3)*8, vsB = (c1 & 3)*8;

  // prologue: stage keys 0..63 into buffer 0
  {
#if HAS_ASYNC_LDS
    async_copy16(Vb + (size_t)vk0*DP_ + vsA, sV + vk0*64 + vsA);
    async_copy16(Vb + (size_t)vk1*DP_ + vsB, sV + vk1*64 + vsB);
    wait_async0();
#else
    *(uint4*)(sV + vk0*64 + vsA) = *(const uint4*)(Vb + (size_t)vk0*DP_ + vsA);
    *(uint4*)(sV + vk1*64 + vsB) = *(const uint4*)(Vb + (size_t)vk1*DP_ + vsB);
#endif
  }
  __syncthreads();

  v8f cacc = {0.f,0.f,0.f,0.f,0.f,0.f,0.f,0.f};
  for (int it = 0; it < 32; ++it) {   // 64 keys per step
    const int cur = it & 1, nxt = cur ^ 1;
    const bool hs = (it + 1 < 32);
    const size_t gnext = (size_t)(it + 1)*64;

#if HAS_ASYNC_LDS
    if (hs) {
      async_copy16(Vb + (gnext + vk0)*DP_ + vsA, sV + nxt*4096 + vk0*64 + vsA);
      async_copy16(Vb + (gnext + vk1)*DP_ + vsB, sV + nxt*4096 + vk1*64 + vsB);
    }
#else
    U16x8 t0, t1;
    if (hs) {
      t0.u = *(const uint4*)(Vb + (gnext + vk0)*DP_ + vsA);
      t1.u = *(const uint4*)(Vb + (gnext + vk1)*DP_ + vsB);
    }
#endif

    // compute on buffer cur: this wave handles keys [it*64 + kh2*32, +32)
    {
      const int kbase = it*64 + kh2*32;
      const _Float16* sVc = sV + cur*4096 + kh2*32*64;
      v16h a, b;
      #pragma unroll
      for (int j = 0; j < 8; ++j) {   // A = probs (f32 in LDS -> f16)
        int k0 = lh*16 + 2*j;
        float2 pv = *(const float2*)(sS + l16*S_ + kbase + k0);
        a[2*j]   = (_Float16)pv.x;
        a[2*j+1] = (_Float16)pv.y;
      }
      const int d = dt*16 + l16;
      #pragma unroll
      for (int j = 0; j < 8; ++j) {   // B = V rows (un-transposed tile)
        int k0 = lh*16 + 2*j;
        b[2*j]   = sVc[k0*64 + d];
        b[2*j+1] = sVc[(k0 + 1)*64 + d];
      }
      cacc = __builtin_amdgcn_wmma_f32_16x16x32_f16(
          false, a, false, b, (short)0, cacc, false, false);
    }

#if HAS_ASYNC_LDS
    if (hs) wait_async0();
#else
    if (hs) {
      *(uint4*)(sV + nxt*4096 + vk0*64 + vsA) = t0.u;
      *(uint4*)(sV + nxt*4096 + vk1*64 + vsB) = t1.u;
    }
#endif
    __syncthreads();
  }

  // reduce the two key-half partials per d-tile (waves w and w+4), then store
  float* sPart = sS;   // probs no longer needed
  if (wave >= 4) {
    #pragma unroll
    for (int r = 0; r < 8; ++r)
      sPart[dt*256 + (lh*8 + r)*16 + l16] = cacc[r];
  }
  __syncthreads();
  if (wave < 4) {
    const int bb = bh >> 4, h = bh & 15;
    #pragma unroll
    for (int r = 0; r < 8; ++r) {
      float val = cacc[r] + sPart[dt*256 + (lh*8 + r)*16 + l16];
      int m = lh*8 + r;
      int srow = qb*16 + m;
      int d = dt*16 + l16;
      ctxOut[((size_t)bb*S_ + srow)*D_ + h*DP_ + d] = (_Float16)val;
    }
  }
}

// ---------------------------------------------------------------- launch
extern "C" void kernel_launch(void* const* d_in, const int* in_sizes, int n_in,
                              void* d_out, int out_size, void* d_ws, size_t ws_size,
                              hipStream_t stream) {
  const float* q  = (const float*)d_in[0];
  const float* k  = (const float*)d_in[1];
  const float* v  = (const float*)d_in[2];
  const float* Wq = (const float*)d_in[3];
  const float* bq = (const float*)d_in[4];
  const float* Wk = (const float*)d_in[5];
  const float* bk = (const float*)d_in[6];
  const float* Wv = (const float*)d_in[7];
  const float* bv = (const float*)d_in[8];
  const float* Wo = (const float*)d_in[9];
  const float* bo = (const float*)d_in[10];

  char* ws = (char*)d_ws;
  const size_t ACT = (size_t)M_*D_*2;   // 8 MB each f16 activation
  const size_t WSZ = (size_t)D_*D_*2;   // 2 MB each f16 weight
  _Float16* qh   = (_Float16*)(ws);
  _Float16* kh   = (_Float16*)(ws + ACT);
  _Float16* vh   = (_Float16*)(ws + 2*ACT);
  _Float16* wq16 = (_Float16*)(ws + 3*ACT);
  _Float16* wk16 = (_Float16*)(ws + 3*ACT + WSZ);
  _Float16* wv16 = (_Float16*)(ws + 3*ACT + 2*WSZ);
  _Float16* wo16 = (_Float16*)(ws + 3*ACT + 3*WSZ);
  _Float16* Qp   = (_Float16*)(ws + 3*ACT + 4*WSZ);
  _Float16* Kp   = (_Float16*)(ws + 4*ACT + 4*WSZ);
  _Float16* Vp   = (_Float16*)(ws + 5*ACT + 4*WSZ);
  _Float16* ctx  = (_Float16*)(ws + 6*ACT + 4*WSZ);  // total 64 MB

  float* outO  = (float*)d_out;
  float* attnO = outO + (size_t)M_*D_;

  const int nA = M_*D_, nW = D_*D_;
  cvt_f32_f16_kernel<<<(nA+255)/256, 256, 0, stream>>>(q,  qh,   nA);
  cvt_f32_f16_kernel<<<(nA+255)/256, 256, 0, stream>>>(k,  kh,   nA);
  cvt_f32_f16_kernel<<<(nA+255)/256, 256, 0, stream>>>(v,  vh,   nA);
  cvt_f32_f16_kernel<<<(nW+255)/256, 256, 0, stream>>>(Wq, wq16, nW);
  cvt_f32_f16_kernel<<<(nW+255)/256, 256, 0, stream>>>(Wk, wk16, nW);
  cvt_f32_f16_kernel<<<(nW+255)/256, 256, 0, stream>>>(Wv, wv16, nW);
  cvt_f32_f16_kernel<<<(nW+255)/256, 256, 0, stream>>>(Wo, wo16, nW);

  dim3 gg(D_/64, M_/128);   // (16, 32)
  gemm_f16_wmma_kernel<0><<<gg, 256, 0, stream>>>(qh, wq16, bq, Qp, nullptr, M_, D_, D_);
  gemm_f16_wmma_kernel<0><<<gg, 256, 0, stream>>>(kh, wk16, bk, Kp, nullptr, M_, D_, D_);
  gemm_f16_wmma_kernel<0><<<gg, 256, 0, stream>>>(vh, wv16, bv, Vp, nullptr, M_, D_, D_);

  (void)hipFuncSetAttribute(reinterpret_cast<const void*>(attn_fused_kernel),
                            hipFuncAttributeMaxDynamicSharedMemorySize,
                            ATTN_SMEM_BYTES);
  attn_fused_kernel<<<dim3(B_*H_*(S_/16)), 256, ATTN_SMEM_BYTES, stream>>>(
      Qp, Kp, Vp, attnO, ctx);

  gemm_f16_wmma_kernel<1><<<gg, 256, 0, stream>>>(ctx, wo16, bo, nullptr, outO, M_, D_, D_);
}
